// FusedAttention_52604759441491
// MI455X (gfx1250) — matrix operand order repaired and checked by reading, not compiled, not run on hardware
//
#include <hip/hip_runtime.h>
#include <math.h>

typedef __bf16 bf16;
typedef __attribute__((ext_vector_type(16))) __bf16 v16bf;
typedef __attribute__((ext_vector_type(8)))  __bf16 v8bf;
typedef __attribute__((ext_vector_type(8)))  float  v8f;

#define DIMN 4096
#define SEQ  2048
#define BATCH 2
#define NH   32
#define NKV  8
#define HD   128
#define QKV_OUT 6144

static __device__ __forceinline__ bf16 f2bf(float f) { return (bf16)f; }
static __device__ __forceinline__ float bf2f(bf16 b) { return (float)b; }

static __device__ __forceinline__ v8f zero8() {
  v8f z;
  #pragma unroll
  for (int i = 0; i < 8; ++i) z[i] = 0.0f;
  return z;
}
static __device__ __forceinline__ v16bf cat8(v8bf lo, v8bf hi) {
  return __builtin_shufflevector(lo, hi, 0, 1, 2, 3, 4, 5, 6, 7,
                                 8, 9, 10, 11, 12, 13, 14, 15);
}
static __device__ __forceinline__ v16bf cvt16(v8f lo, v8f hi) {
  v8bf l8 = __builtin_convertvector(lo, v8bf);   // v_cvt_pk_bf16_f32 x4
  v8bf h8 = __builtin_convertvector(hi, v8bf);
  return cat8(l8, h8);
}
static __device__ __forceinline__ v8f wmma_bf16(v16bf a, v16bf b, v8f c) {
  // D = A(16x32 bf16) * B(32x16 bf16) + C(16x16 f32)
  return __builtin_amdgcn_wmma_f32_16x16x32_bf16(false, a, false, b, (short)0, c,
                                                 false, false);
}

// ---------------------------------------------------------------------------
// Stage 1: QKV = x @ w_qkv^T  (M=B*S=4096, N=6144, K=4096), bf16 WMMA.
// Block tile 128x128, 8 waves (4x2), wave tile 32x64.
// Ping-pong double-buffered k-loop (64 per iteration) -> zero register
// rotation, loads of one buffer overlap MMAs on the other.
// Epilogue scatters into Q[b,h,s,d], K[b,kv,s,d], Vt[b,kv,d,s] (bf16).
// ---------------------------------------------------------------------------
__global__ __launch_bounds__(256)
void qkv_gemm(const float* __restrict__ x, const float* __restrict__ w,
              bf16* __restrict__ Qb, bf16* __restrict__ Kb, bf16* __restrict__ Vt) {
  const int lane = threadIdx.x & 31;
  const int wave = threadIdx.x >> 5;
  const int m  = lane & 15;
  const int hi = lane >> 4;
  const int kA = hi * 8;
  const int kB = hi * 8;
  const int rowBase = blockIdx.y * 128 + (wave >> 1) * 32;
  const int colBase = blockIdx.x * 128 + (wave & 1) * 64;

  const float* apBase[2];
  #pragma unroll
  for (int mf = 0; mf < 2; ++mf)
    apBase[mf] = x + (size_t)(rowBase + mf * 16 + m) * DIMN + kA;
  const float* wpBase[4];
  #pragma unroll
  for (int nf = 0; nf < 4; ++nf)
    wpBase[nf] = w + (size_t)(colBase + nf * 16 + m) * DIMN + kB;

  v8f acc[2][4];
  #pragma unroll
  for (int mf = 0; mf < 2; ++mf)
    #pragma unroll
    for (int nf = 0; nf < 4; ++nf) acc[mf][nf] = zero8();

  auto loadTile = [&](int kk, v16bf* a, v16bf* bq) {
    #pragma unroll
    for (int mf = 0; mf < 2; ++mf)
      a[mf] = cvt16(*(const v8f*)(apBase[mf] + kk),
                    *(const v8f*)(apBase[mf] + kk + 16));
    #pragma unroll
    for (int nf = 0; nf < 4; ++nf)
      bq[nf] = cvt16(*(const v8f*)(wpBase[nf] + kk),
                     *(const v8f*)(wpBase[nf] + kk + 16));
  };
  auto mma = [&](const v16bf* a, const v16bf* bq) {
    #pragma unroll
    for (int mf = 0; mf < 2; ++mf)
      #pragma unroll
      for (int nf = 0; nf < 4; ++nf)
        acc[mf][nf] = wmma_bf16(a[mf], bq[nf], acc[mf][nf]);
  };

  v16bf a0[2], b0[4], a1[2], b1[4];
  loadTile(0, a0, b0);
  for (int kk = 0; kk < DIMN; kk += 64) {
    loadTile(kk + 32, a1, b1);                    // fill pong
    __builtin_prefetch(apBase[0] + kk + 128, 0, 1);
    __builtin_prefetch(wpBase[0] + kk + 128, 0, 1);
    mma(a0, b0);                                  // consume ping
    if (kk + 64 < DIMN) loadTile(kk + 64, a0, b0);// refill ping
    mma(a1, b1);                                  // consume pong
  }

  // epilogue: region branch is wave-uniform (colU is 16-aligned, regions
  // 128-aligned) -> scalar branches, no exec masking
  #pragma unroll
  for (int mf = 0; mf < 2; ++mf) {
    #pragma unroll
    for (int nf = 0; nf < 4; ++nf) {
      const int colU = colBase + nf * 16;
      if (colU < NH * HD) {                       // Q
        const int h = colU >> 7, d = (colU & 127) + m;
        #pragma unroll
        for (int r = 0; r < 8; ++r) {
          const int row = rowBase + mf * 16 + r + 8 * hi;
          const int b = row >> 11, s = row & (SEQ - 1);
          Qb[(((size_t)b * NH + h) * SEQ + s) * HD + d] = f2bf(acc[mf][nf][r]);
        }
      } else if (colU < (NH + NKV) * HD) {        // K
        const int c2 = colU - NH * HD;
        const int h = c2 >> 7, d = (c2 & 127) + m;
        #pragma unroll
        for (int r = 0; r < 8; ++r) {
          const int row = rowBase + mf * 16 + r + 8 * hi;
          const int b = row >> 11, s = row & (SEQ - 1);
          Kb[(((size_t)b * NKV + h) * SEQ + s) * HD + d] = f2bf(acc[mf][nf][r]);
        }
      } else {                                    // V (transposed: [d][s])
        const int c2 = colU - (NH + NKV) * HD;
        const int h = c2 >> 7, d = (c2 & 127) + m;
        #pragma unroll
        for (int r = 0; r < 8; ++r) {
          const int row = rowBase + mf * 16 + r + 8 * hi;
          const int b = row >> 11, s = row & (SEQ - 1);
          Vt[(((size_t)b * NKV + h) * HD + d) * SEQ + s] = f2bf(acc[mf][nf][r]);
        }
      }
    }
  }
}

// ---------------------------------------------------------------------------
// Stage 2: RoPE in place on Q and K (bf16). Pair (i, i+64), i in [0,64).
// ---------------------------------------------------------------------------
__global__ void rope_kernel(bf16* __restrict__ Qb, bf16* __restrict__ Kb) {
  const long long QP = (long long)BATCH * NH * SEQ * 64;
  const long long KP = (long long)BATCH * NKV * SEQ * 64;
  long long idx = (long long)blockIdx.x * blockDim.x + threadIdx.x;
  bf16* buf; long long t;
  if (idx < QP)            { buf = Qb; t = idx; }
  else if (idx < QP + KP)  { buf = Kb; t = idx - QP; }
  else return;
  const int i = (int)(t & 63);
  const int s = (int)((t >> 6) & (SEQ - 1));
  const long long bh = t >> 17;                    // SEQ*64 = 2^17
  const size_t base = ((size_t)bh * SEQ + s) * HD;
  // inv_freq = ROPE_BASE^(-i/64), ln(500000) = 13.122363377
  const float inv_freq = expf(-(float)i * (13.122363377404328f / 64.0f));
  const float ang = (float)s * inv_freq;
  float sn, cs;
  sincosf(ang, &sn, &cs);
  const float x1 = bf2f(buf[base + i]);
  const float x2 = bf2f(buf[base + i + 64]);
  buf[base + i]      = f2bf(x1 * cs - x2 * sn);
  buf[base + i + 64] = f2bf(x2 * cs + x1 * sn);
}

// ---------------------------------------------------------------------------
// Stage 3: causal GQA flash attention. One wave per (b, h, 16-query tile).
// Key tiles of 32. QK^T: 8 wmma / tile. P via LDS relayout. P@V: 8 wmma / tile.
// ---------------------------------------------------------------------------
__global__ __launch_bounds__(32)
void attn_kernel(const bf16* __restrict__ Q, const bf16* __restrict__ K,
                 const bf16* __restrict__ Vt, bf16* __restrict__ O) {
  __shared__ __align__(64) bf16 Pld[16][32];

  const int lane = threadIdx.x;
  const int bid  = blockIdx.x;
  const int qt = bid & 127;            // S/16 = 128 tiles
  const int h  = (bid >> 7) & 31;
  const int b  = bid >> 12;
  const int q0 = qt * 16;
  const int kv = h >> 2;               // N_REP = 4
  const int m  = lane & 15;
  const int hi = lane >> 4;
  const int kA = hi * 16;
  const int kB = hi * 16;

  const bf16* Qp = Q + (((size_t)b * NH + h) * SEQ + q0) * HD + (size_t)m * HD;
  const bf16* Kbase = K + ((size_t)b * NKV + kv) * SEQ * HD;
  const bf16* Vbase = Vt + ((size_t)b * NKV + kv) * HD * SEQ;

  // Q A-fragments for d = 0,32,64,96
  v16bf qf[4];
  #pragma unroll
  for (int dsi = 0; dsi < 4; ++dsi) {
    v8bf lo = *(const v8bf*)(Qp + dsi * 32 + kA);
    v8bf hv = *(const v8bf*)(Qp + dsi * 32 + 8 + kA);
    qf[dsi] = cat8(lo, hv);
  }

  v8f acc[8];
  #pragma unroll
  for (int nf = 0; nf < 8; ++nf) acc[nf] = zero8();
  float mrow[8], lrow[8];
  #pragma unroll
  for (int r = 0; r < 8; ++r) { mrow[r] = -INFINITY; lrow[r] = 0.0f; }

  const float scale = 0.08838834764831845f;  // 1/sqrt(128)

  for (int k0 = 0; k0 <= q0 + 15; k0 += 32) {
    // ---- scores S = Q K^T for 32 keys (two 16-wide fragments) ----
    v8f sf[2];
    #pragma unroll
    for (int f = 0; f < 2; ++f) {
      sf[f] = zero8();
      const int key = k0 + f * 16 + m;           // B col n = lane&15
      const bf16* kp = Kbase + (size_t)key * HD + kB;
      #pragma unroll
      for (int dsi = 0; dsi < 4; ++dsi) {
        v16bf bb = *(const v16bf*)(kp + dsi * 32);
        sf[f] = wmma_bf16(qf[dsi], bb, sf[f]);
      }
    }

    // ---- mask + scale + online softmax stats ----
    #pragma unroll
    for (int r = 0; r < 8; ++r) {
      const int qrow = q0 + r + 8 * hi;
      #pragma unroll
      for (int f = 0; f < 2; ++f) {
        const int keyc = k0 + f * 16 + m;
        float sv = sf[f][r];
        sf[f][r] = (keyc <= qrow) ? sv * scale : -1e30f;
      }
      float t = fmaxf(sf[0][r], sf[1][r]);
      t = fmaxf(t, __shfl_xor(t, 1));
      t = fmaxf(t, __shfl_xor(t, 2));
      t = fmaxf(t, __shfl_xor(t, 4));
      t = fmaxf(t, __shfl_xor(t, 8));
      const float mn = fmaxf(mrow[r], t);
      const float alpha = __expf(mrow[r] - mn);
      mrow[r] = mn;
      const float p0 = __expf(sf[0][r] - mn);
      const float p1 = __expf(sf[1][r] - mn);
      sf[0][r] = p0; sf[1][r] = p1;
      float rs = p0 + p1;
      rs += __shfl_xor(rs, 1);
      rs += __shfl_xor(rs, 2);
      rs += __shfl_xor(rs, 4);
      rs += __shfl_xor(rs, 8);
      lrow[r] = lrow[r] * alpha + rs;
      #pragma unroll
      for (int nf = 0; nf < 8; ++nf) acc[nf][r] *= alpha;
    }

    // ---- relayout P (C-layout f32) -> A-layout bf16 via LDS ----
    __syncthreads();
    #pragma unroll
    for (int f = 0; f < 2; ++f)
      #pragma unroll
      for (int r = 0; r < 8; ++r)
        Pld[r + 8 * hi][f * 16 + m] = f2bf(sf[f][r]);
    __syncthreads();
    v8bf plo = *(const v8bf*)(&Pld[m][kA]);
    v8bf phi = *(const v8bf*)(&Pld[m][8 + kA]);
    v16bf aP = cat8(plo, phi);

    // ---- acc += P @ V  (8 dim-tiles of 16) ----
    #pragma unroll
    for (int nf = 0; nf < 8; ++nf) {
      const bf16* vp = Vbase + (size_t)(nf * 16 + m) * SEQ + k0 + kB;
      v16bf bv = *(const v16bf*)vp;
      acc[nf] = wmma_bf16(aP, bv, acc[nf]);
    }
    __syncthreads();
  }

  // ---- normalize and write O[b, s, h*128 + d] (bf16) ----
  #pragma unroll
  for (int nf = 0; nf < 8; ++nf) {
    const int col = h * HD + nf * 16 + m;
    #pragma unroll
    for (int r = 0; r < 8; ++r) {
      const int row = q0 + r + 8 * hi;
      const float v = acc[nf][r] / lrow[r];
      O[((size_t)b * SEQ + row) * DIMN + col] = f2bf(v);
    }
  }
}

// ---------------------------------------------------------------------------
// Stage 4: out = O @ w_o^T  (M=4096, N=4096, K=4096), fp32 output,
// ping-pong double-buffered k-loop, A already bf16.
// ---------------------------------------------------------------------------
__global__ __launch_bounds__(256)
void out_gemm(const bf16* __restrict__ O, const float* __restrict__ w,
              float* __restrict__ out) {
  const int lane = threadIdx.x & 31;
  const int wave = threadIdx.x >> 5;
  const int m  = lane & 15;
  const int hi = lane >> 4;
  const int kA = hi * 8;
  const int kB = hi * 8;
  const int rowBase = blockIdx.y * 128 + (wave >> 1) * 32;
  const int colBase = blockIdx.x * 128 + (wave & 1) * 64;

  const bf16* apBase[2];
  #pragma unroll
  for (int mf = 0; mf < 2; ++mf)
    apBase[mf] = O + (size_t)(rowBase + mf * 16 + m) * DIMN + kA;
  const float* wpBase[4];
  #pragma unroll
  for (int nf = 0; nf < 4; ++nf)
    wpBase[nf] = w + (size_t)(colBase + nf * 16 + m) * DIMN + kB;

  v8f acc[2][4];
  #pragma unroll
  for (int mf = 0; mf < 2; ++mf)
    #pragma unroll
    for (int nf = 0; nf < 4; ++nf) acc[mf][nf] = zero8();

  auto loadTile = [&](int kk, v16bf* a, v16bf* bq) {
    #pragma unroll
    for (int mf = 0; mf < 2; ++mf)
      a[mf] = cat8(*(const v8bf*)(apBase[mf] + kk),
                   *(const v8bf*)(apBase[mf] + kk + 16));
    #pragma unroll
    for (int nf = 0; nf < 4; ++nf)
      bq[nf] = cvt16(*(const v8f*)(wpBase[nf] + kk),
                     *(const v8f*)(wpBase[nf] + kk + 16));
  };
  auto mma = [&](const v16bf* a, const v16bf* bq) {
    #pragma unroll
    for (int mf = 0; mf < 2; ++mf)
      #pragma unroll
      for (int nf = 0; nf < 4; ++nf)
        acc[mf][nf] = wmma_bf16(a[mf], bq[nf], acc[mf][nf]);
  };

  v16bf a0[2], b0[4], a1[2], b1[4];
  loadTile(0, a0, b0);
  for (int kk = 0; kk < DIMN; kk += 64) {
    loadTile(kk + 32, a1, b1);
    __builtin_prefetch(apBase[0] + kk + 128, 0, 1);
    __builtin_prefetch(wpBase[0] + kk + 128, 0, 1);
    mma(a0, b0);
    if (kk + 64 < DIMN) loadTile(kk + 64, a0, b0);
    mma(a1, b1);
  }

  #pragma unroll
  for (int mf = 0; mf < 2; ++mf)
    #pragma unroll
    for (int nf = 0; nf < 4; ++nf) {
      const int col = colBase + nf * 16 + m;
      #pragma unroll
      for (int r = 0; r < 8; ++r) {
        const int row = rowBase + mf * 16 + r + 8 * hi;
        out[(size_t)row * DIMN + col] = acc[mf][nf][r];
      }
    }
}

// ---------------------------------------------------------------------------
extern "C" void kernel_launch(void* const* d_in, const int* in_sizes, int n_in,
                              void* d_out, int out_size, void* d_ws, size_t ws_size,
                              hipStream_t stream) {
  const float* x     = (const float*)d_in[0];
  const float* w_qkv = (const float*)d_in[1];
  const float* w_o   = (const float*)d_in[2];
  float* out = (float*)d_out;

  // bf16 workspace: Q (33.5MB) | K (8.4MB) | Vt (8.4MB) | O (33.5MB) = 80MB
  bf16* Qb = (bf16*)d_ws;
  bf16* Kb = Qb + (size_t)BATCH * NH * SEQ * HD;
  bf16* Vt = Kb + (size_t)BATCH * NKV * SEQ * HD;
  bf16* O  = Vt + (size_t)BATCH * NKV * SEQ * HD;

  dim3 g1(QKV_OUT / 128, (BATCH * SEQ) / 128);
  qkv_gemm<<<g1, 256, 0, stream>>>(x, w_qkv, Qb, Kb, Vt);

  const long long pairs = (long long)BATCH * (NH + NKV) * SEQ * 64;
  rope_kernel<<<(int)((pairs + 255) / 256), 256, 0, stream>>>(Qb, Kb);

  attn_kernel<<<BATCH * NH * (SEQ / 16), 32, 0, stream>>>(Qb, Kb, Vt, O);

  dim3 g4(DIMN / 128, (BATCH * SEQ) / 128);
  out_gemm<<<g4, 256, 0, stream>>>(O, w_o, out);
}
